// AttnConvKernel_80204219285683
// MI455X (gfx1250) — matrix operand order repaired and checked by reading, compiled, not run on hardware
//
#include <hip/hip_runtime.h>
#include <hip/hip_bf16.h>

// CDNA5 / gfx1250, wave32. Matrix math via v_wmma_f32_16x16x32_bf16;
// LDS staging via global_load_async_to_lds_b128 + s_wait_asynccnt (ASYNCcnt).
//
// Pipeline:
//  k0 prep_weights : w1,w2 f32 -> Wp bf16 [9][384co][128ci]   (tap-major)
//  k1 conv_wmma    : implicit-GEMM conv (M=384, K=9*128, N=pixels),
//                    4h x 48w spatial tiles; WMMA N-subtiles are same-phase
//                    (w stride 3) so Y stores are coalesced 32B runs;
//                    weight slabs double-buffered via async-to-LDS;
//                    output phase-major bf16: Y[b][k][c][l]
//  k2 attn_wmma    : per (b,k,z of 8): partial S[co][ci] = Y2.Y1 over 2048 l,
//                    async-to-LDS double-buffered chunks, partials -> ws
//  k3 softmax_rows : fold 8 partials (fixed order), scale, softmax -> d_out

typedef __attribute__((ext_vector_type(16))) __bf16 v16bf;
typedef __attribute__((ext_vector_type(8)))  __bf16 v8bf;
typedef __attribute__((ext_vector_type(8)))  float  v8f;

#define NB    4      // batch
#define CI    128    // input channels
#define COT   384    // fused out channels (y1:0..127, y2:128..383)
#define HW    384    // H == W
#define LPH   16384  // pixels per phase = (384/3)^2
#define XROW  136    // padded ci stride in conv LDS rows (272B, 16B-aligned)
#define XCOL  50     // conv LDS cols: 48 + 2 halo
#define KPAD  136    // padded l stride in attn LDS rows (272B, 16B-aligned)
#define KSPL  8      // attn K-split
#define PS    1179648  // partial-buffer stride (= out elements)

static __device__ __forceinline__ v16bf make16(v8bf lo, v8bf hi) {
  return __builtin_shufflevector(lo, hi,
      0,1,2,3,4,5,6,7,8,9,10,11,12,13,14,15);
}

// ---- CDNA5 async LDS staging -----------------------------------------------
static __device__ __forceinline__ void async_load_b128(const void* lds_dst,
                                                       const void* gsrc) {
  unsigned loff = (unsigned)(size_t)lds_dst;          // LDS byte offset
  unsigned long long ga = (unsigned long long)(size_t)gsrc;
  asm volatile("global_load_async_to_lds_b128 %0, %1, off"
               :: "v"(loff), "v"(ga) : "memory");
}
static __device__ __forceinline__ void wait_asynccnt0() {
  asm volatile("s_wait_asynccnt 0x0" ::: "memory");
}

// ---------------------------------------------------------------------------
// k0: weights -> bf16, layout Wp[kk][co][ci], kk = kh*3+kw, co in [0,384)
// ---------------------------------------------------------------------------
__global__ __launch_bounds__(256) void prep_weights(
    const float* __restrict__ w1, const float* __restrict__ w2,
    __bf16* __restrict__ Wp)
{
  int idx = blockIdx.x * 256 + threadIdx.x;
  if (idx >= 9 * COT * CI) return;
  int ci = idx % CI;
  int t  = idx / CI;
  int co = t % COT;
  int kk = t / COT;
  float v = (co < CI) ? w1[(co * CI + ci) * 9 + kk]
                      : w2[((co - CI) * CI + ci) * 9 + kk];
  Wp[((size_t)kk * COT + co) * CI + ci] = (__bf16)v;
}

// ---------------------------------------------------------------------------
// k1: conv as implicit GEMM. Block = 128 out-channels x (4h x 48w) pixels.
//     N-subtile sn covers phase sn: pixels w = w0 + sn + 3*lane (w0 % 3 == 0).
//     K loop = 9 taps x 4 ci-chunks of 32. Weight slabs async double-buffered.
// ---------------------------------------------------------------------------
__global__ __launch_bounds__(256) void conv_wmma(
    const float* __restrict__ x, const __bf16* __restrict__ Wp,
    __bf16* __restrict__ Y)
{
  const int mTile = blockIdx.x;            // co block: 0..2  (128 each)
  int strip = blockIdx.y;                  // ((b*96 + hB)*8 + wB)
  const int wB = strip % 8;  strip /= 8;
  const int hB = strip % 96;
  const int b  = strip / 96;
  const int h0 = hB * 4;
  const int w0 = wB * 48;                  // divisible by 3

  const int tid  = threadIdx.x;
  const int lane = tid & 31;
  const int wave = tid >> 5;
  const int wm   = wave & 1;               // 64-row block of C (co)
  const int wn   = wave >> 1;              // h-row 0..3 of the spatial tile
  const int lhalf = (lane >> 4);           // 0/1: K-group select
  const int ln16  = lane & 15;

  extern __shared__ char smem[];
  __bf16* sX  = (__bf16*)smem;                                  // [6][50][XROW]
  __bf16* sWb = (__bf16*)(smem + (size_t)6 * XCOL * XROW * 2);  // [2][128][XROW]

  // async issue of one 128co x 128ci weight slab into buffer `buf`
  auto issueW = [&](int buf, int kk) {
    const __bf16* wsrc = Wp + ((size_t)kk * COT + mTile * 128) * CI;
    __bf16* dst = sWb + (size_t)buf * 128 * XROW;
    for (int idx = tid; idx < 128 * 16; idx += 256) {   // 2048 = 8*256
      int row = idx >> 4, seg = idx & 15;
      async_load_b128(&dst[row * XROW + seg * 8], wsrc + row * CI + seg * 8);
    }
  };

  issueW(0, 0);  // DMA runs under the x conversion below

  // ---- x neighborhood (6 rows x 50 cols x 128 ci), f32->bf16, ci-contig ----
  const int totX = 6 * CI * XCOL;          // 38400 = 150 * 256 (uniform)
  for (int idx = tid; idx < totX; idx += 256) {
    int col = idx % XCOL;
    int t   = idx / XCOL;
    int ci  = t % CI;
    int r   = t / CI;
    int gh = h0 + r - 1;
    int gw = w0 + col - 1;
    float v = 0.0f;
    if ((unsigned)gh < (unsigned)HW && (unsigned)gw < (unsigned)HW)
      v = x[(((size_t)b * CI + ci) * HW + gh) * HW + gw];
    sX[(r * XCOL + col) * XROW + ci] = (__bf16)v;
  }

  v8f acc[4][3] = {};

  int cw = 0;
  for (int kk = 0; kk < 9; ++kk) {
    const int kh = kk / 3, kw = kk % 3;
    wait_asynccnt0();
    __syncthreads();                 // slab(cw) ready; readers of cw^1 retired
    if (kk < 8) issueW(cw ^ 1, kk + 1);
    const __bf16* sW = sWb + (size_t)cw * 128 * XROW;

    for (int cb = 0; cb < CI; cb += 32) {
      // A: 16x32 bf16; lane<16: K 0..7,16..23 ; lane>=16: 8..15,24..31
      v16bf afr[4];
#pragma unroll
      for (int sm = 0; sm < 4; ++sm) {
        int co  = wm * 64 + sm * 16 + ln16;
        int ci0 = cb + lhalf * 8;
        const __bf16* p = &sW[co * XROW + ci0];
        afr[sm] = make16(*(const v8bf*)p, *(const v8bf*)(p + 16));
      }
      // B: 32x16 bf16; lane = pixel column (phase sn, w stride 3)
      v16bf bfr[3];
#pragma unroll
      for (int sn = 0; sn < 3; ++sn) {
        int c   = sn + 3 * ln16;              // local w within 48-wide tile
        int ci0 = cb + lhalf * 8;
        const __bf16* q = &sX[((wn + kh) * XCOL + c + kw) * XROW + ci0];
        bfr[sn] = make16(*(const v8bf*)q, *(const v8bf*)(q + 16));
      }
#pragma unroll
      for (int sm = 0; sm < 4; ++sm)
#pragma unroll
        for (int sn = 0; sn < 3; ++sn)
          acc[sm][sn] = __builtin_amdgcn_wmma_f32_16x16x32_bf16(
              false, afr[sm], false, bfr[sn], (short)0, acc[sm][sn],
              false, false);
    }
    cw ^= 1;
  }

  // ---- store phase-major bf16: Y[b][k][c][l]; per sn: kph uniform,
  //      l = (h/3)*128 + wB*16 + lane -> 16 consecutive l per 16 lanes.
  const int h = h0 + wn;
  const int kh3 = h % 3, ph = h / 3;
  const int l = ph * 128 + wB * 16 + ln16;
#pragma unroll
  for (int sn = 0; sn < 3; ++sn) {
    int kph = kh3 * 3 + sn;
    __bf16* yb = Y + (((size_t)b * 9 + kph) * COT) * LPH + l;
#pragma unroll
    for (int sm = 0; sm < 4; ++sm) {
#pragma unroll
      for (int r = 0; r < 8; ++r) {
        int co = mTile * 128 + wm * 64 + sm * 16 + r + lhalf * 8;
        yb[(size_t)co * LPH] = (__bf16)acc[sm][sn][r];
      }
    }
  }
}

// ---------------------------------------------------------------------------
// k2: per (b,k,z): partial S[co][ci] = sum_{l in split z} Y2[co][l]*Y1[ci][l]
//     Kc=128 chunks, async-to-LDS double buffered. Partials (unscaled) -> Sp.
// ---------------------------------------------------------------------------
__global__ __launch_bounds__(256) void attn_wmma(
    const __bf16* __restrict__ Y, float* __restrict__ Sp)
{
  const int mTile = blockIdx.x;            // co block: 0..1 (128 each)
  const int bk    = blockIdx.y;            // 0..35
  const int z     = blockIdx.z;            // 0..KSPL-1 (K split)
  const int b = bk / 9, kph = bk % 9;
  const int kbeg = z * (LPH / KSPL);
  const int kend = kbeg + (LPH / KSPL);

  const __bf16* Y2 = Y + (((size_t)b * 9 + kph) * COT + CI + mTile * 128) * LPH;
  const __bf16* Y1 = Y + (((size_t)b * 9 + kph) * COT) * LPH;

  const int tid  = threadIdx.x;
  const int lane = tid & 31;
  const int wave = tid >> 5;
  const int wm   = wave & 1;
  const int wn   = wave >> 1;
  const int lhalf = (lane >> 4);
  const int ln16  = lane & 15;

  extern __shared__ char smem[];
  const size_t MATE = (size_t)128 * KPAD;            // elems per matrix buffer
  __bf16* sA = (__bf16*)smem;                        // [2][128][KPAD]
  __bf16* sB = (__bf16*)smem + 2 * MATE;             // [2][128][KPAD]

  // async issue of one 128x128 bf16 chunk of A (Y2) and B (Y1)
  auto issueAB = [&](int buf, int k0) {
    __bf16* da = sA + (size_t)buf * MATE;
    __bf16* db = sB + (size_t)buf * MATE;
    for (int idx = tid; idx < 128 * 16; idx += 256) {  // 2048 = 8*256
      int row = idx >> 4, seg = idx & 15;
      const size_t go = (size_t)row * LPH + k0 + seg * 8;
      async_load_b128(&da[row * KPAD + seg * 8], Y2 + go);
      async_load_b128(&db[row * KPAD + seg * 8], Y1 + go);
    }
  };

  v8f acc[4][2] = {};

  issueAB(0, kbeg);
  int cur = 0;
  for (int k0 = kbeg; k0 < kend; k0 += 128) {
    wait_asynccnt0();
    __syncthreads();               // chunk(cur) ready; readers of cur^1 retired
    if (k0 + 128 < kend) issueAB(cur ^ 1, k0 + 128);
    const __bf16* cA = sA + (size_t)cur * MATE;
    const __bf16* cB = sB + (size_t)cur * MATE;

    for (int cb = 0; cb < 128; cb += 32) {
      v16bf afr[4];
#pragma unroll
      for (int sm = 0; sm < 4; ++sm) {
        int co = wm * 64 + sm * 16 + ln16;
        int l0 = cb + lhalf * 8;
        const __bf16* p = &cA[co * KPAD + l0];
        afr[sm] = make16(*(const v8bf*)p, *(const v8bf*)(p + 16));
      }
      v16bf bfr[2];
#pragma unroll
      for (int sn = 0; sn < 2; ++sn) {
        int ci = wn * 32 + sn * 16 + ln16;
        int l0 = cb + lhalf * 8;
        const __bf16* q = &cB[ci * KPAD + l0];
        bfr[sn] = make16(*(const v8bf*)q, *(const v8bf*)(q + 16));
      }
#pragma unroll
      for (int sm = 0; sm < 4; ++sm)
#pragma unroll
        for (int sn = 0; sn < 2; ++sn)
          acc[sm][sn] = __builtin_amdgcn_wmma_f32_16x16x32_bf16(
              false, afr[sm], false, bfr[sn], (short)0, acc[sm][sn],
              false, false);
    }
    cur ^= 1;
  }

  // raw partial sums -> Sp[z][b][co(256)][ci(128)][9]
#pragma unroll
  for (int sn = 0; sn < 2; ++sn) {
    int ci = wn * 32 + sn * 16 + ln16;
#pragma unroll
    for (int sm = 0; sm < 4; ++sm) {
#pragma unroll
      for (int r = 0; r < 8; ++r) {
        int co = mTile * 128 + wm * 64 + sm * 16 + r + lhalf * 8;
        Sp[(size_t)z * PS + (((size_t)b * 256 + co) * CI + ci) * 9 + kph] =
            acc[sm][sn][r];
      }
    }
  }
}

// ---------------------------------------------------------------------------
// k3: fold KSPL partials (fixed order), scale, softmax over rows of 1152.
// ---------------------------------------------------------------------------
__global__ __launch_bounds__(256) void softmax_rows(
    const float* __restrict__ Sp, float* __restrict__ out)
{
  const int row = blockIdx.x;                // b*256 + co
  const float scale = 0.029462782549439483f; // 1/sqrt(128*9)
  float* p = out + (size_t)row * 1152;
  const float* q = Sp + (size_t)row * 1152;
  const int tid = threadIdx.x;
  const int lane = tid & 31, wave = tid >> 5;
  __shared__ float red[8];

  float m = -3.0e38f;
  for (int i = tid; i < 1152; i += 256) {
    float v = 0.0f;
#pragma unroll
    for (int zz = 0; zz < KSPL; ++zz) v += q[i + (size_t)zz * PS];
    v *= scale;
    p[i] = v;
    m = fmaxf(m, v);
  }
  for (int off = 16; off > 0; off >>= 1) m = fmaxf(m, __shfl_xor(m, off));
  if (lane == 0) red[wave] = m;
  __syncthreads();
  float mm = red[0];
#pragma unroll
  for (int i = 1; i < 8; ++i) mm = fmaxf(mm, red[i]);
  __syncthreads();

  float s = 0.0f;
  for (int i = tid; i < 1152; i += 256) {
    float e = expf(p[i] - mm);
    p[i] = e;
    s += e;
  }
  for (int off = 16; off > 0; off >>= 1) s += __shfl_xor(s, off);
  if (lane == 0) red[wave] = s;
  __syncthreads();
  float tot = 0.0f;
#pragma unroll
  for (int i = 0; i < 8; ++i) tot += red[i];
  float inv = 1.0f / tot;
  for (int i = tid; i < 1152; i += 256) p[i] *= inv;
}

// ---------------------------------------------------------------------------
extern "C" void kernel_launch(void* const* d_in, const int* in_sizes, int n_in,
                              void* d_out, int out_size, void* d_ws,
                              size_t ws_size, hipStream_t stream)
{
  (void)in_sizes; (void)n_in; (void)out_size; (void)ws_size;
  const float* x  = (const float*)d_in[0];   // [4,128,384,384]
  const float* w1 = (const float*)d_in[1];   // [128,128,3,3]
  const float* w2 = (const float*)d_in[2];   // [256,128,3,3]
  float* out = (float*)d_out;                // [4,256,128,9]

  // scratch: Wp (0.9MB @0) | Y bf16 phase-major (432MB @1MB) | partials (@433MB)
  __bf16* Wp = (__bf16*)d_ws;
  __bf16* Yp = (__bf16*)((char*)d_ws + ((size_t)1 << 20));
  float*  Sp = (float*)((char*)d_ws + ((size_t)433 << 20));

  prep_weights<<<(9 * COT * CI + 255) / 256, 256, 0, stream>>>(w1, w2, Wp);

  const size_t ldsConv = ((size_t)6 * XCOL * XROW + (size_t)2 * 128 * XROW) * 2;
  conv_wmma<<<dim3(3, NB * 96 * 8), 256, ldsConv, stream>>>(x, Wp, Yp);

  const size_t ldsAttn = (size_t)4 * 128 * KPAD * 2;
  attn_wmma<<<dim3(2, NB * 9, KSPL), 256, ldsAttn, stream>>>(Yp, Sp);

  softmax_rows<<<NB * 256, 256, 0, stream>>>(Sp, out);
}
